// STU_11450382811252
// MI455X (gfx1250) — matrix operand order repaired and checked by reading, not compiled, not run on hardware
//
#include <hip/hip_runtime.h>
#include <hip/hip_bf16.h>

// ---------------------------------------------------------------------------
// STU forward on MI455X (gfx1250, wave32, WMMA + TDM).
//
// delta_phi as a lower-triangular block-Toeplitz matmul:
//   delta_phi[t,o] = sum_s sum_k (v[s,k]*sigma_k^0.25) * Z[t-s,k,o],
//   Z[t,k,o] = sum_c x[t,c]*m_phi[k*64+c,o]
// Projection and Toeplitz contraction use v_wmma_f32_16x16x32_f16.
// Filter (A-operand) tiles are DMA'd into LDS with tensor_load_to_lds,
// double-buffered on TENSORcnt. Working set (~107 MB) fits in the 192 MB L2.
// ---------------------------------------------------------------------------

typedef __attribute__((ext_vector_type(16))) _Float16 v16h;
typedef __attribute__((ext_vector_type(8)))  float    v8f;
typedef unsigned int u32;
typedef __attribute__((ext_vector_type(4))) unsigned int u32x4;
typedef __attribute__((ext_vector_type(8))) int i32x8;
typedef __attribute__((ext_vector_type(4))) int i32x4;

#define BATCH 8
#define SEQL  4096
#define DDIM  64
#define KEIG  24
#define KU    3
#define TBLK  16
#define NBLK  (SEQL / TBLK)        // 256 time blocks
#define KRED  (TBLK * KEIG)        // 384 reduction per block pair
#define NCH   (KRED / 32)          // 12 WMMA k-steps per block pair
#define FTILE (NCH * 32)           // v16h elements per filter tile (12 KB)

// Workspace layout (bytes)
#define F_BYTES   ((size_t)NBLK * NCH * 32 * 16 * 2)              // 3,145,728
#define MB_BYTES  ((size_t)2 * 96 * 32 * 16 * 2)                  //   196,608
#define ZT_BYTES  ((size_t)BATCH * NBLK * NCH * 4 * 32 * 16 * 2)  // 100,663,296

// WMMA A-operand (16x32 f16) element -> K mapping for a lane's kgroup.
__device__ __forceinline__ int a_kk(int e, int kgrp) {
  return kgrp * 16 + e;
}

// TDM: 1-D DMA of `nbytes` (multiple of 4) from global into LDS.
// D# built per CDNA5 ISA: group0 = {flags, lds_addr,
// global_addr, type=2}; group1 = dims/strides with data_size=4B.
__device__ __forceinline__ void tdm_load_tile(u32 lds_off, const void* gptr,
                                              u32 nbytes) {
  unsigned long long ga = (unsigned long long)(uintptr_t)gptr;
  u32 ndw = nbytes >> 2;
  u32x4 g0;
  g0[0] = 1u;                                           // count=1, no gather
  g0[1] = lds_off;                                      // LDS byte address
  g0[2] = (u32)(ga & 0xffffffffu);                      // global addr [31:0]
  g0[3] = (u32)((ga >> 32) & 0x1ffffffu) | (2u << 30);  // addr [56:32] | type=2
  i32x8 g1;
  g1[0] = (int)(2u << 16);                    // data_size=4B, mask/pad/iter=0
  g1[1] = (int)((ndw & 0xffffu) << 16);       // tensor_dim0[15:0] @ bits63:48
  g1[2] = (int)(((ndw >> 16) & 0xffffu) | (1u << 16)); // dim0 hi | tensor_dim1=1
  g1[3] = (int)((ndw & 0xffffu) << 16);       // tile_dim0 @ bits127:112
  g1[4] = 0;                                  // tile_dim1=0, tile_dim2=0
  g1[5] = (int)ndw;                           // tensor_dim0_stride low32
  g1[6] = 0;
  g1[7] = 0;
  i32x4 z4 = {0, 0, 0, 0};
#if __clang_major__ >= 23
  i32x8 z8 = {0, 0, 0, 0, 0, 0, 0, 0};
  __builtin_amdgcn_tensor_load_to_lds(g0, g1, z4, z4, z8, 0);
#else
  __builtin_amdgcn_tensor_load_to_lds(g0, g1, z4, z4, 0);
#endif
}

// --- K1a: filter tiles F[dI][ch][lane][16] in WMMA-A layout.
__global__ void stu_build_filters(const float* __restrict__ eig_vals,
                                  const float* __restrict__ eig_vecs,
                                  _Float16* __restrict__ F) {
  int dI   = blockIdx.x;
  int tid  = threadIdx.x;           // 0..383 = (ch, lane)
  int ch   = tid >> 5;
  int lane = tid & 31;
  int i    = lane & 15;
  int kgrp = lane >> 4;
  _Float16* dst = F + (((size_t)dI * NCH + ch) * 32 + lane) * 16;
#pragma unroll
  for (int e = 0; e < 16; ++e) {
    int kk = a_kk(e, kgrp);
    int r  = ch * 32 + kk;          // r = j*KEIG + k
    int j  = r / KEIG;
    int k  = r - j * KEIG;
    int s  = TBLK * dI + i - j;     // causal filter tap
    float v = 0.f;
    if (s >= 0 && s < SEQL)
      v = eig_vecs[(size_t)s * KEIG + k] * powf(eig_vals[k], 0.25f);
    dst[e] = (_Float16)v;
  }
}

// --- K1b: m_phi in WMMA-B layout for the Z projection GEMM.
__global__ void stu_build_mphi(const float* __restrict__ m_phi,
                               _Float16* __restrict__ Mb) {
  int blk    = blockIdx.x;          // (cchunk, ntile)
  int cchunk = blk / 96;
  int ntile  = blk - cchunk * 96;
  int lane   = threadIdx.x;
  int n      = lane & 15;
  int khalf  = lane >> 4;
  _Float16* dst = Mb + (((size_t)cchunk * 96 + ntile) * 32 + lane) * 16;
  int q = ntile * 16 + n;
  int k = q >> 6, o = q & 63;
#pragma unroll
  for (int e = 0; e < 16; ++e) {
    int c = cchunk * 32 + khalf * 16 + e;
    dst[e] = (_Float16)m_phi[((size_t)k * 64 + c) * 64 + o];
  }
}

// --- K2b: deltas initialized with the AR-u term.
__global__ void stu_ar_init(const float* __restrict__ x,
                            const float* __restrict__ m_u,
                            float* __restrict__ deltas) {
  size_t idx = (size_t)blockIdx.x * blockDim.x + threadIdx.x;
  int o = (int)(idx & 63);
  size_t bt = idx >> 6;
  int t = (int)(bt & (SEQL - 1));
  int b = (int)(bt >> 12);
  float s = 0.f;
#pragma unroll
  for (int k = 0; k < KU; ++k) {
    if (t >= k) {
      const float* xr = x + ((size_t)b * SEQL + (t - k)) * DDIM;
#pragma unroll 8
      for (int i = 0; i < DDIM; ++i)
        s += m_u[((size_t)o * DDIM + i) * KU + k] * xr[i];
    }
  }
  deltas[idx] = s;
}

// --- K2: Z projection via WMMA, scattered into WMMA-B layout.
__global__ void stu_project(const float* __restrict__ x,
                            const _Float16* __restrict__ Mb,
                            _Float16* __restrict__ Zt) {
  int J = blockIdx.x, b = blockIdx.y;
  int lane = threadIdx.x;           // one wave
  int i = lane & 15, kgrp = lane >> 4;
  const float* xr = x + ((size_t)b * SEQL + (J * TBLK + i)) * DDIM;
  v16h a0, a1;
#pragma unroll
  for (int e = 0; e < 16; ++e) {
    a0[e] = (_Float16)xr[a_kk(e, kgrp)];
    a1[e] = (_Float16)xr[32 + a_kk(e, kgrp)];
  }
  const v16h* Bp = reinterpret_cast<const v16h*>(Mb);
  size_t zbase = ((size_t)b * NBLK + J) * ((size_t)NCH * 4 * 32 * 16);
  int n = lane & 15;
  int mh = lane >> 4;
  for (int nt = 0; nt < 96; ++nt) {
    v8f acc = {};
    v16h b0 = Bp[(size_t)(0 * 96 + nt) * 32 + lane];
    v16h b1 = Bp[(size_t)(1 * 96 + nt) * 32 + lane];
    acc = __builtin_amdgcn_wmma_f32_16x16x32_f16(false, a0, false, b0,
                                                 (short)0, acc, false, false);
    acc = __builtin_amdgcn_wmma_f32_16x16x32_f16(false, a1, false, b1,
                                                 (short)0, acc, false, false);
#pragma unroll
    for (int g = 0; g < 8; ++g) {
      int j = g + mh * 8;           // C-layout row = time within block
      int q = nt * 16 + n;          // q = k*64 + o
      int k = q >> 6, o = q & 63;
      int r = j * KEIG + k;         // B-operand row for the Toeplitz GEMM
      int ch = r >> 5, kk = r & 31;
      int ot = o >> 4, n2 = o & 15;
      int lane2 = n2 + ((kk >> 4) << 4);
      Zt[zbase + (((size_t)ch * 4 + ot) * 32 + lane2) * 16 + (kk & 15)] =
          (_Float16)acc[g];
    }
  }
}

// --- K3: main block-Toeplitz contraction. A (filter) tiles are DMA'd into
// LDS by the Tensor Data Mover, double-buffered on TENSORcnt; each of the
// 4 waves owns one 16-wide o-tile and streams its B operand from L2.
__global__ void __launch_bounds__(128)
stu_toeplitz(const _Float16* __restrict__ F,
             const _Float16* __restrict__ Zt,
             float* __restrict__ deltas) {
  __shared__ v16h smA[2][FTILE];    // 2 x 12 KB double buffer
  int I = blockIdx.x, b = blockIdx.y;
  int lane = threadIdx.x & 31;
  int wv   = threadIdx.x >> 5;      // o-tile 0..3
  const v16h* Zp = reinterpret_cast<const v16h*>(Zt) +
                   ((size_t)b * NBLK * NCH * 4 + wv) * 32 + lane;
  v8f acc = {};
  if (wv == 0)                      // prefetch A tile for J=0 (dI = I)
    tdm_load_tile((u32)(uintptr_t)&smA[0][0],
                  F + (size_t)I * FTILE * 16, FTILE * 32u);
  for (int J = 0; J <= I; ++J) {
    int pb = J & 1;
    if (wv == 0) {
      if (J < I) {
        tdm_load_tile((u32)(uintptr_t)&smA[pb ^ 1][0],
                      F + (size_t)(I - J - 1) * FTILE * 16, FTILE * 32u);
        __builtin_amdgcn_s_wait_tensorcnt(1);   // tile J landed, J+1 in flight
      } else {
        __builtin_amdgcn_s_wait_tensorcnt(0);
      }
    }
    __syncthreads();                // hand A(J) to all waves
    const v16h* Bp = Zp + (size_t)J * (NCH * 4 * 32);
#pragma unroll
    for (int ch = 0; ch < NCH; ++ch) {
      v16h av = smA[pb][ch * 32 + lane];        // ds_load_b128 x2
      v16h bv = Bp[(size_t)ch * 4 * 32];        // global b128 x2 (L2-resident)
      acc = __builtin_amdgcn_wmma_f32_16x16x32_f16(false, av, false, bv,
                                                   (short)0, acc, false, false);
    }
    __syncthreads();                // all waves done before buffer reuse
  }
  int n = lane & 15, mh = lane >> 4;
#pragma unroll
  for (int g = 0; g < 8; ++g) {
    int t = I * TBLK + g + mh * 8;
    deltas[((size_t)b * SEQL + t) * DDIM + wv * 16 + n] += acc[g];
  }
}

// --- K4: order-2 matrix recurrence, one workgroup per batch.
__global__ void stu_recurrence(const float* __restrict__ deltas,
                               const float* __restrict__ m_y,
                               float* __restrict__ out) {
  int b = blockIdx.x;
  int o = threadIdx.x;              // 64 threads = 2 waves
  __shared__ float y1[DDIM], y2[DDIM];
  float a1[DDIM], a2[DDIM];
#pragma unroll 8
  for (int i = 0; i < DDIM; ++i) {
    a1[i] = m_y[(size_t)o * 2 * DDIM + i];
    a2[i] = m_y[(size_t)o * 2 * DDIM + DDIM + i];
  }
  y1[o] = 0.f; y2[o] = 0.f;
  __syncthreads();
  for (int t = 0; t < SEQL; ++t) {
    float s = deltas[((size_t)b * SEQL + t) * DDIM + o];
#pragma unroll 8
    for (int i = 0; i < DDIM; ++i)
      s += a1[i] * y1[i] + a2[i] * y2[i];
    __syncthreads();
    y2[o] = y1[o];
    y1[o] = s;
    out[((size_t)b * SEQL + t) * DDIM + o] = s;
    __syncthreads();
  }
}

extern "C" void kernel_launch(void* const* d_in, const int* in_sizes, int n_in,
                              void* d_out, int out_size, void* d_ws, size_t ws_size,
                              hipStream_t stream) {
  const float* x       = (const float*)d_in[0];   // [8,4096,64]
  const float* m_y     = (const float*)d_in[1];   // [64,2,64]
  const float* m_u     = (const float*)d_in[2];   // [64,64,3]
  const float* m_phi   = (const float*)d_in[3];   // [1536,64]
  const float* eigvals = (const float*)d_in[4];   // [24]
  const float* eigvecs = (const float*)d_in[5];   // [4096,24]
  char* ws = (char*)d_ws;
  _Float16* F      = (_Float16*)(ws);
  _Float16* Mb     = (_Float16*)(ws + F_BYTES);
  _Float16* Zt     = (_Float16*)(ws + F_BYTES + MB_BYTES);
  float*    deltas = (float*)   (ws + F_BYTES + MB_BYTES + ZT_BYTES);
  float*    out    = (float*)d_out;

  stu_build_filters<<<NBLK, NCH * 32, 0, stream>>>(eigvals, eigvecs, F);
  stu_build_mphi<<<2 * 96, 32, 0, stream>>>(m_phi, Mb);
  stu_ar_init<<<(BATCH * SEQL * DDIM) / 256, 256, 0, stream>>>(x, m_u, deltas);
  dim3 g2(NBLK, BATCH);
  stu_project<<<g2, 32, 0, stream>>>(x, Mb, Zt);
  stu_toeplitz<<<g2, 128, 0, stream>>>(F, Zt, deltas);
  stu_recurrence<<<BATCH, DDIM, 0, stream>>>(deltas, m_y, out);
}
